// CrossAttention_38723425140909
// MI455X (gfx1250) — matrix operand order repaired and checked by reading, not compiled, not run on hardware
//
#include <hip/hip_runtime.h>

typedef __attribute__((ext_vector_type(16))) __bf16 v16bf;
typedef __attribute__((ext_vector_type(8)))  __bf16 v8bf;
typedef __attribute__((ext_vector_type(8)))  float  v8f;
typedef unsigned short u16;

#define B_FRAMES 16
#define N_TOK    640
#define C_DIM    640
#define H_HEADS  8
#define D_HEAD   80
#define NSEL     2048
#define M_ROWS   (B_FRAMES * N_TOK)   /* 10240 */
#define KV_LEN   (N_TOK + NSEL)       /* 2688  */

// ---------------- helpers ----------------
__device__ __forceinline__ u16 f2bf_raw(float f) {
  unsigned u = __float_as_uint(f);
  unsigned r = u + 0x7FFFu + ((u >> 16) & 1u);   // round-to-nearest-even
  return (u16)(r >> 16);
}

// ---------------- kernel 1: SACFA mask compaction ----------------
__global__ void sacfa_compact(const float* __restrict__ mask, int* __restrict__ sel) {
  __shared__ int cnt;
  if (threadIdx.x == 0) cnt = 0;
  __syncthreads();
  for (int i = threadIdx.x; i < M_ROWS; i += blockDim.x) {
    if (mask[i] > 0.5f) {
      int p = atomicAdd(&cnt, 1);
      sel[p] = i;
    }
  }
}

// ---------------- kernel 2: f32 -> bf16 convert ----------------
__global__ void cvt_f32_bf16(const float* __restrict__ src, u16* __restrict__ dst, int n) {
  for (int i = blockIdx.x * blockDim.x + threadIdx.x; i < n; i += gridDim.x * blockDim.x)
    dst[i] = f2bf_raw(src[i]);
}

// ---------------- kernel 3: weight transpose+convert: Wt[n,k] = bf16(W[k,n]) ----------------
__global__ void xpose_w(const float* __restrict__ w, u16* __restrict__ wt) {
  int i = blockIdx.x * blockDim.x + threadIdx.x;   // over 640*640
  if (i < C_DIM * C_DIM) {
    int k = i / C_DIM, n = i % C_DIM;
    wt[(size_t)n * C_DIM + k] = f2bf_raw(w[i]);
  }
}

// ---------------- kernel 4: gather+transpose V: Vt[b][d][j] over kv sequence ----------------
__global__ void pack_vt(const u16* __restrict__ Vb, const int* __restrict__ sel,
                        u16* __restrict__ Vt) {
  const size_t per_b = (size_t)C_DIM * KV_LEN;
  size_t i = (size_t)blockIdx.x * blockDim.x + threadIdx.x;   // (b, d, j), j fastest
  if (i >= (size_t)B_FRAMES * per_b) return;
  const int b = (int)(i / per_b);
  const size_t r = i % per_b;
  const int d = (int)(r / KV_LEN);
  const int j = (int)(r % KV_LEN);
  const int tok = (j < N_TOK) ? (b * N_TOK + j) : sel[j - N_TOK];
  Vt[i] = Vb[(size_t)tok * C_DIM + d];   // writes coalesced over j
}

// ---------------- kernel 5: bf16 WMMA GEMM, 32x32 register tile per wave ----------------
__global__ void gemm_bf16_wmma(const u16* __restrict__ Ar,   // bf16 [M,640] row-major
                               const u16* __restrict__ Wtr,  // bf16 [640(n),640(k)]
                               u16* __restrict__ outb,       // bf16 out (or null)
                               float* __restrict__ outf,     // f32 out (or null)
                               const float* __restrict__ bias) {
  const int lane = threadIdx.x & 31;
  const int wid  = threadIdx.x >> 5;
  const int tile = blockIdx.x * 8 + wid;     // over (M/32)*(N/32) = 6400
  const int NT2  = C_DIM / 32;               // 20
  const int mt   = tile / NT2;
  const int nt   = tile % NT2;
  const int r16  = lane & 15;
  const int hi   = (lane >= 16) ? 1 : 0;

  const __bf16* A  = reinterpret_cast<const __bf16*>(Ar);
  const __bf16* Wt = reinterpret_cast<const __bf16*>(Wtr);
  const __bf16* arow0 = A  + (size_t)(mt * 32 + r16) * C_DIM;
  const __bf16* arow1 = arow0 + (size_t)16 * C_DIM;
  const __bf16* brow0 = Wt + (size_t)(nt * 32 + r16) * C_DIM;
  const __bf16* brow1 = brow0 + (size_t)16 * C_DIM;

  union AV { v16bf v; v8bf p[2]; };

  v8f acc00 = {}, acc01 = {}, acc10 = {}, acc11 = {};
#pragma unroll 2
  for (int kb = 0; kb < C_DIM; kb += 32) {
    __builtin_prefetch(arow0 + kb + 256, 0, 1);
    AV a0, a1;
    a0.p[0] = *(const v8bf*)(arow0 + kb + hi * 16);
    a0.p[1] = *(const v8bf*)(arow0 + kb + hi * 16 + 8);
    a1.p[0] = *(const v8bf*)(arow1 + kb + hi * 16);
    a1.p[1] = *(const v8bf*)(arow1 + kb + hi * 16 + 8);
    v16bf b0 = *(const v16bf*)(brow0 + kb + hi * 16);
    v16bf b1 = *(const v16bf*)(brow1 + kb + hi * 16);
    acc00 = __builtin_amdgcn_wmma_f32_16x16x32_bf16(false, a0.v, false, b0, (short)0, acc00, false, false);
    acc01 = __builtin_amdgcn_wmma_f32_16x16x32_bf16(false, a0.v, false, b1, (short)0, acc01, false, false);
    acc10 = __builtin_amdgcn_wmma_f32_16x16x32_bf16(false, a1.v, false, b0, (short)0, acc10, false, false);
    acc11 = __builtin_amdgcn_wmma_f32_16x16x32_bf16(false, a1.v, false, b1, (short)0, acc11, false, false);
  }

  const int col0 = nt * 32 + r16;
  const int col1 = col0 + 16;
  const float bv0 = bias ? bias[col0] : 0.0f;
  const float bv1 = bias ? bias[col1] : 0.0f;
#pragma unroll
  for (int r = 0; r < 8; r++) {
    const size_t row0 = (size_t)(mt * 32 + r + hi * 8);
    const size_t row1 = row0 + 16;
    if (outf) {
      outf[row0 * C_DIM + col0] = acc00[r] + bv0;
      outf[row0 * C_DIM + col1] = acc01[r] + bv1;
      outf[row1 * C_DIM + col0] = acc10[r] + bv0;
      outf[row1 * C_DIM + col1] = acc11[r] + bv1;
    } else {
      outb[row0 * C_DIM + col0] = f2bf_raw(acc00[r]);
      outb[row0 * C_DIM + col1] = f2bf_raw(acc01[r]);
      outb[row1 * C_DIM + col0] = f2bf_raw(acc10[r]);
      outb[row1 * C_DIM + col1] = f2bf_raw(acc11[r]);
    }
  }
}

// ---------------- kernel 6: flash attention over 2688 kv tokens ----------------
// One wave handles one (b, h, 16-query-row) tile; kv processed in steps of 32.
__global__ __launch_bounds__(256) void flash_attn(const u16* __restrict__ Qr,
                                                  const u16* __restrict__ Kr,
                                                  const u16* __restrict__ VtR,
                                                  const int* __restrict__ sel,
                                                  u16* __restrict__ Or) {
  __shared__ __align__(64) int rowidx_s[8][32];
  __shared__ __align__(64) u16 sP[8][16][32];     // P tile, bf16 raw bits

  const int lane = threadIdx.x & 31;
  const int wid  = threadIdx.x >> 5;
  const int gt   = blockIdx.x * 8 + wid;          // 0..5119
  const int qt   = gt % 40;
  const int h    = (gt / 40) % H_HEADS;
  const int b    = gt / (40 * H_HEADS);

  const int r16 = lane & 15;
  const int hi  = (lane >= 16) ? 1 : 0;

  const __bf16* Q  = reinterpret_cast<const __bf16*>(Qr);
  const __bf16* K  = reinterpret_cast<const __bf16*>(Kr);
  const __bf16* Vt = reinterpret_cast<const __bf16*>(VtR);

  union AV { v16bf v; v8bf p[2]; };
  const v8bf  z8  = {};
  const v16bf z16 = {};

  // ---- Q fragments (A layout), d = [0,32) [32,64) [64,96-with-zero-pad) ----
  const __bf16* qp = Q + (size_t)(b * N_TOK + qt * 16 + r16) * C_DIM + h * D_HEAD;
  AV aq[3];
#pragma unroll
  for (int c = 0; c < 2; c++) {
    aq[c].p[0] = *(const v8bf*)(qp + c * 32 + hi * 16);
    aq[c].p[1] = *(const v8bf*)(qp + c * 32 + hi * 16 + 8);
  }
  aq[2].p[0] = hi ? z8 : *(const v8bf*)(qp + 64);   // d 64..71 (hi half: 80..87 padded)
  aq[2].p[1] = hi ? z8 : *(const v8bf*)(qp + 72);   // d 72..79 (hi half: 88..95 padded)

  // V^T row base for this (b, h): Vt[b][d][kv], row stride KV_LEN
  const __bf16* vtb = Vt + ((size_t)b * C_DIM + h * D_HEAD) * KV_LEN;

  const float scale = 0.11180339887498949f;        // 80^-0.5
  v8f o[5];
#pragma unroll
  for (int t = 0; t < 5; t++) o[t] = (v8f){};
  float m[8], lsum[8];
#pragma unroll
  for (int r = 0; r < 8; r++) { m[r] = -1e30f; lsum[r] = 0.0f; }

  for (int kv0 = 0; kv0 < KV_LEN; kv0 += 32) {
    // stage the 32 kv-row indices (gathered rows come from sel[]) for the K gather
    {
      int p   = kv0 + lane;
      int row = (p < N_TOK) ? (b * N_TOK + p) : sel[p - N_TOK];
      rowidx_s[wid][lane] = row;
    }
    asm volatile("s_wait_dscnt 0x0" ::: "memory");

    // ---- scores S[16 x 32] via 6 WMMA (two 16-col tiles, 3 d-chunks each) ----
    v8f s0 = {}, s1 = {};
    {
      const int tok0 = rowidx_s[wid][r16];
      const int tok1 = rowidx_s[wid][16 + r16];
      const __bf16* kp0 = K + (size_t)tok0 * C_DIM + h * D_HEAD;
      const __bf16* kp1 = K + (size_t)tok1 * C_DIM + h * D_HEAD;
#pragma unroll
      for (int c = 0; c < 3; c++) {
        v16bf b0, b1;
        if (c < 2) {
          b0 = *(const v16bf*)(kp0 + c * 32 + hi * 16);
          b1 = *(const v16bf*)(kp1 + c * 32 + hi * 16);
        } else {
          b0 = hi ? z16 : *(const v16bf*)(kp0 + 64);
          b1 = hi ? z16 : *(const v16bf*)(kp1 + 64);
        }
        s0 = __builtin_amdgcn_wmma_f32_16x16x32_bf16(false, aq[c].v, false, b0,
                                                     (short)0, s0, false, false);
        s1 = __builtin_amdgcn_wmma_f32_16x16x32_bf16(false, aq[c].v, false, b1,
                                                     (short)0, s1, false, false);
      }
    }

    // ---- online softmax (row reductions across the 16-lane half) ----
#pragma unroll
    for (int r = 0; r < 8; r++) {
      float v0 = s0[r] * scale, v1 = s1[r] * scale;
      float mx = fmaxf(v0, v1);
      mx = fmaxf(mx, __shfl_xor(mx, 1, 32));
      mx = fmaxf(mx, __shfl_xor(mx, 2, 32));
      mx = fmaxf(mx, __shfl_xor(mx, 4, 32));
      mx = fmaxf(mx, __shfl_xor(mx, 8, 32));
      float mn   = fmaxf(m[r], mx);
      float corr = __expf(m[r] - mn);
      float p0   = __expf(v0 - mn);
      float p1   = __expf(v1 - mn);
      float rs   = p0 + p1;
      rs += __shfl_xor(rs, 1, 32);
      rs += __shfl_xor(rs, 2, 32);
      rs += __shfl_xor(rs, 4, 32);
      rs += __shfl_xor(rs, 8, 32);
      lsum[r] = lsum[r] * corr + rs;
      m[r]    = mn;
#pragma unroll
      for (int t = 0; t < 5; t++) o[t][r] *= corr;
      const int prow = r + hi * 8;
      sP[wid][prow][r16]      = f2bf_raw(p0);
      sP[wid][prow][16 + r16] = f2bf_raw(p1);
    }
    asm volatile("s_wait_dscnt 0x0" ::: "memory");

    // ---- P as A-fragment (transpose through LDS) ----
    AV ap;
    ap.p[0] = *(const v8bf*)((const __bf16*)&sP[wid][r16][hi * 16]);
    ap.p[1] = *(const v8bf*)((const __bf16*)&sP[wid][r16][hi * 16 + 8]);

    // ---- O += P x V : 5 WMMA, each B-fragment is ONE contiguous v16bf load from Vt ----
#pragma unroll
    for (int t = 0; t < 5; t++) {
      const __bf16* vp = vtb + (size_t)(t * 16 + r16) * KV_LEN + kv0 + hi * 16;
      v16bf vb = *(const v16bf*)vp;
      o[t] = __builtin_amdgcn_wmma_f32_16x16x32_bf16(false, ap.v, false, vb,
                                                     (short)0, o[t], false, false);
    }
  }

  // ---- epilogue: normalize and store bf16 ----
#pragma unroll
  for (int r = 0; r < 8; r++) {
    const float inv  = 1.0f / lsum[r];
    const size_t row = (size_t)(b * N_TOK + qt * 16 + r + hi * 8);
#pragma unroll
    for (int t = 0; t < 5; t++)
      Or[row * C_DIM + h * D_HEAD + t * 16 + r16] = f2bf_raw(o[t][r] * inv);
  }
}

// ---------------- launcher ----------------
extern "C" void kernel_launch(void* const* d_in, const int* in_sizes, int n_in,
                              void* d_out, int out_size, void* d_ws, size_t ws_size,
                              hipStream_t stream) {
  const float* x    = (const float*)d_in[0];
  const float* Wq   = (const float*)d_in[1];
  const float* Wk   = (const float*)d_in[2];
  const float* Wv   = (const float*)d_in[3];
  const float* Wo   = (const float*)d_in[4];
  const float* bo   = (const float*)d_in[5];
  const float* mask = (const float*)d_in[6];
  (void)in_sizes; (void)n_in; (void)out_size; (void)ws_size;

  constexpr size_t SZ_ACT = (size_t)M_ROWS * C_DIM * sizeof(u16);           // 13,107,200
  constexpr size_t SZ_W   = (size_t)C_DIM * C_DIM * sizeof(u16);            //    819,200
  constexpr size_t SZ_VT  = (size_t)B_FRAMES * C_DIM * KV_LEN * sizeof(u16);// 55,050,240
  char* ws = (char*)d_ws;
  int* sel = (int*)ws;                              // 2048 ints
  size_t off = 8192;
  u16* xb  = (u16*)(ws + off); off += SZ_ACT;
  u16* wqt = (u16*)(ws + off); off += SZ_W;
  u16* wkt = (u16*)(ws + off); off += SZ_W;
  u16* wvt = (u16*)(ws + off); off += SZ_W;
  u16* wot = (u16*)(ws + off); off += SZ_W;
  u16* Qb  = (u16*)(ws + off); off += SZ_ACT;
  u16* Kb  = (u16*)(ws + off); off += SZ_ACT;
  u16* Vb  = (u16*)(ws + off); off += SZ_ACT;
  u16* Ab  = (u16*)(ws + off); off += SZ_ACT;
  u16* Vt  = (u16*)(ws + off); off += SZ_VT;        // total ~124 MB, L2-resident

  // 1. mask compaction
  sacfa_compact<<<1, 1024, 0, stream>>>(mask, sel);
  // 2. convert x to bf16
  cvt_f32_bf16<<<2048, 256, 0, stream>>>(x, xb, M_ROWS * C_DIM);
  // 3. transpose+convert weights
  xpose_w<<<1600, 256, 0, stream>>>(Wq, wqt);
  xpose_w<<<1600, 256, 0, stream>>>(Wk, wkt);
  xpose_w<<<1600, 256, 0, stream>>>(Wv, wvt);
  xpose_w<<<1600, 256, 0, stream>>>(Wo, wot);
  // 4. QKV projections (WMMA, 32x32 tiles): 6400 wave-tiles, 8 waves/block
  gemm_bf16_wmma<<<800, 256, 0, stream>>>(xb, wqt, Qb, nullptr, nullptr);
  gemm_bf16_wmma<<<800, 256, 0, stream>>>(xb, wkt, Kb, nullptr, nullptr);
  gemm_bf16_wmma<<<800, 256, 0, stream>>>(xb, wvt, Vb, nullptr, nullptr);
  // 5. gather + transpose V into Vt[b][d][kv]
  pack_vt<<<(int)((SZ_VT / sizeof(u16) + 255) / 256), 256, 0, stream>>>(Vb, sel, Vt);
  // 6. flash attention: 5120 wave-tiles
  flash_attn<<<640, 256, 0, stream>>>(Qb, Kb, Vt, sel, Ab);
  // 7. output projection + bias -> f32 d_out
  gemm_bf16_wmma<<<800, 256, 0, stream>>>(Ab, wot, nullptr, (float*)d_out, bo);
}
